// MoELayer_30872224924369
// MI455X (gfx1250) — matrix operand order repaired and checked by reading, not compiled, not run on hardware
//
#include <hip/hip_runtime.h>
#include <hip/hip_bf16.h>

// ---------------- problem constants ----------------
#define T_TOK   4096          // B*S tokens
#define DDIM    768
#define IDIM    3072
#define NEXP    8
#define TOPK    2
#define MAXSLOTS 9216         // 4096*2 routed pairs + 8 experts * 127 pad, rounded
#define MAXMBLK  72           // MAXSLOTS / 128

typedef __attribute__((ext_vector_type(16))) __bf16    v16bf;
typedef __attribute__((ext_vector_type(8)))  float     v8f;
typedef __attribute__((ext_vector_type(4)))  unsigned  u32x4;

union FragBF { u32x4 q[2]; v16bf v; };

__device__ __forceinline__ unsigned short f2bf(float f) {
    unsigned int u = __float_as_uint(f);
    u += 0x7fffu + ((u >> 16) & 1u);      // round-to-nearest-even
    return (unsigned short)(u >> 16);
}

// async global -> LDS, 16 bytes per lane (CDNA5 GLOBAL_LOAD_ASYNC_TO_LDS_B128)
__device__ __forceinline__ void async_g2l_b128(unsigned lds_off, const void* gaddr) {
    asm volatile("global_load_async_to_lds_b128 %0, %1, off"
                 :: "v"(lds_off), "v"(gaddr) : "memory");
}

// ---------------- 1. weight transpose + f32->bf16 ----------------
// dst[e][n][k] = bf16(src[e][k][n]);  dst row (fixed n) is K-contiguous.
__global__ void moe_convertT(const float* __restrict__ src,
                             unsigned short* __restrict__ dst,
                             int N0, int K0) {
    long long i = (long long)blockIdx.x * blockDim.x + threadIdx.x;
    long long total = (long long)NEXP * N0 * K0;
    if (i >= total) return;
    int k = (int)(i % K0);
    long long r = i / K0;
    int n = (int)(r % N0);
    int e = (int)(r / N0);
    dst[i] = f2bf(src[((size_t)e * K0 + k) * (size_t)N0 + n]);
}

// ---------------- 2. router: logits -> softmax -> top-2 ----------------
__global__ void moe_router(const float* __restrict__ x,
                           const float* __restrict__ Wr,
                           const float* __restrict__ br,
                           float* __restrict__ topg,
                           int*   __restrict__ topi) {
    int t    = blockIdx.x;
    int wid  = threadIdx.x >> 5;     // wave == expert
    int lane = threadIdx.x & 31;
    const float* xr = x + (size_t)t * DDIM;
    float s = 0.f;
    for (int d = lane; d < DDIM; d += 32)
        s += xr[d] * Wr[d * NEXP + wid];
    for (int off = 16; off > 0; off >>= 1)
        s += __shfl_xor(s, off);
    __shared__ float lg[NEXP];
    if (lane == 0) lg[wid] = s + br[wid];
    __syncthreads();
    if (threadIdx.x == 0) {
        float m = lg[0];
        for (int e = 1; e < NEXP; e++) m = fmaxf(m, lg[e]);
        float ex[NEXP]; float sum = 0.f;
        for (int e = 0; e < NEXP; e++) { ex[e] = expf(lg[e] - m); sum += ex[e]; }
        int i0 = 0;
        for (int e = 1; e < NEXP; e++) if (ex[e] > ex[i0]) i0 = e;
        int i1 = (i0 == 0) ? 1 : 0;
        for (int e = 0; e < NEXP; e++) if (e != i0 && ex[e] > ex[i1]) i1 = e;
        float inv = 1.f / sum;
        topg[t * 2 + 0] = ex[i0] * inv;  topi[t * 2 + 0] = i0;
        topg[t * 2 + 1] = ex[i1] * inv;  topi[t * 2 + 1] = i1;
    }
}

// ---------------- 3. build per-expert slot lists (single block) ----------------
// meta[0..8]  : expert slot offsets (multiples of 128), meta[8]=total slots
// meta[9]     : number of 128-row M blocks
// meta[10+b]  : expert id of M block b
__global__ void moe_build(const int* __restrict__ topi,
                          int* __restrict__ meta,
                          int* __restrict__ p2t,
                          int* __restrict__ tslot) {
    __shared__ int cnt[NEXP], cnt2[NEXP], off[NEXP + 1];
    int tid = threadIdx.x;
    if (tid < NEXP) { cnt[tid] = 0; cnt2[tid] = 0; }
    __syncthreads();
    for (int i = tid; i < T_TOK * TOPK; i += 256)
        atomicAdd(&cnt[topi[i]], 1);
    __syncthreads();
    if (tid == 0) {
        int o = 0;
        for (int e = 0; e < NEXP; e++) {
            off[e] = o; meta[e] = o;
            int padded = (cnt[e] + 127) & ~127;
            for (int b = 0; b < padded / 128; b++)
                meta[10 + o / 128 + b] = e;
            for (int s = o + cnt[e]; s < o + padded; s++)
                p2t[s] = 0;                         // pad slots -> safe token 0
            o += padded;
        }
        off[NEXP] = o; meta[8] = o; meta[9] = o / 128;
    }
    __syncthreads();
    for (int i = tid; i < T_TOK * TOPK; i += 256) {
        int e = topi[i];
        int pos = atomicAdd(&cnt2[e], 1);
        int s = off[e] + pos;
        p2t[s] = i >> 1;
        tslot[i] = s;
    }
}

// ---------------- 4. gather token rows -> bf16 A matrix ----------------
__global__ void moe_gather(const float* __restrict__ x,
                           const int* __restrict__ p2t,
                           const int* __restrict__ meta,
                           unsigned short* __restrict__ Xg) {
    int s = blockIdx.x;
    if (s >= meta[8]) return;
    const float* src = x + (size_t)p2t[s] * DDIM;
    unsigned short* dst = Xg + (size_t)s * DDIM;
    for (int d = threadIdx.x; d < DDIM; d += 128)
        dst[d] = f2bf(src[d]);
}

// ---------------- 5. block-tiled bf16 WMMA GEMM ----------------
// Block = 256 threads = 8 waves; tile 128M x 64N; K staged 32 wide.
// Double-buffered LDS filled by GLOBAL_LOAD_ASYNC_TO_LDS_B128 (ASYNCcnt),
// so stage k+1 streams in under the WMMAs of stage k.
// A: [rows][Kd] bf16 row-major.  Bw: [E][Nd][Kd] bf16 (K contiguous per column).
__global__ __launch_bounds__(256)
void moe_gemm(const unsigned short* __restrict__ A,
              const unsigned short* __restrict__ Bw,
              const float* __restrict__ bias,
              float* __restrict__ outF,
              unsigned short* __restrict__ outH,
              const int* __restrict__ meta,
              int Kd, int Nd, int doGelu) {
    int nMB = meta[9];
    int by = blockIdx.y;
    if (by >= nMB) return;
    int e     = meta[10 + by];
    int mBase = by * 128;
    int n0    = blockIdx.x * 64;
    const unsigned short* Bx = Bw + (size_t)e * Nd * Kd;

    __shared__ __align__(16) unsigned short As[2][128 * 40]; // 32 cols padded to 40
    __shared__ __align__(16) unsigned short Bs[2][64 * 40];

    int tid = threadIdx.x, wid = tid >> 5, lane = tid & 31;
    int h = lane >> 4, l15 = lane & 15;

    // per-thread staging coordinates (3 async b128 ops per thread per stage)
    int ar0 = tid >> 2, ac = tid & 3;            // A chunk 0: rows 0..63
    int ar1 = ar0 + 64;                          // A chunk 1: rows 64..127
    int br_ = tid >> 2, bc = tid & 3;            // B chunk  : rows(n) 0..63
    unsigned aOff0[2], aOff1[2], bOff[2];
    #pragma unroll
    for (int bdx = 0; bdx < 2; bdx++) {
        aOff0[bdx] = (unsigned)(uintptr_t)(&As[bdx][0] + ar0 * 40 + ac * 8);
        aOff1[bdx] = (unsigned)(uintptr_t)(&As[bdx][0] + ar1 * 40 + ac * 8);
        bOff[bdx]  = (unsigned)(uintptr_t)(&Bs[bdx][0] + br_ * 40 + bc * 8);
    }
    const unsigned short* aSrc0 = A  + (size_t)(mBase + ar0) * Kd + ac * 8;
    const unsigned short* aSrc1 = A  + (size_t)(mBase + ar1) * Kd + ac * 8;
    const unsigned short* bSrc  = Bx + (size_t)(n0 + br_)    * Kd + bc * 8;

    v8f acc[4] = {v8f{}, v8f{}, v8f{}, v8f{}};

    int nK = Kd / 32;
    // prologue: stage 0 into buffer 0
    async_g2l_b128(aOff0[0], aSrc0);
    async_g2l_b128(aOff1[0], aSrc1);
    async_g2l_b128(bOff[0],  bSrc);

    for (int ks = 0; ks < nK; ks++) {
        int cur = ks & 1;
        if (ks + 1 < nK) {
            int k1 = (ks + 1) * 32;
            async_g2l_b128(aOff0[cur ^ 1], aSrc0 + k1);
            async_g2l_b128(aOff1[cur ^ 1], aSrc1 + k1);
            async_g2l_b128(bOff[cur ^ 1],  bSrc  + k1);
            // 3 newest ops = next stage; <=3 outstanding => current stage landed
            asm volatile("s_wait_asynccnt 3" ::: "memory");
        } else {
            asm volatile("s_wait_asynccnt 0" ::: "memory");
        }
        __syncthreads();   // current stage visible workgroup-wide

        // A fragment (16x32 bf16): lane<16 -> K {0..7,16..23}, lane>=16 -> +8
        FragBF a;
        const unsigned short* as = &As[cur][0];
        const unsigned short* bs = &Bs[cur][0];
        a.q[0] = *(const u32x4*)(as + (wid * 16 + l15) * 40 + h * 8);
        a.q[1] = *(const u32x4*)(as + (wid * 16 + l15) * 40 + h * 8 + 16);
        #pragma unroll
        for (int ns = 0; ns < 4; ns++) {
            // B fragment (32x16 bf16): column = lane&15, lane<16 -> K {0..7,16..23}, lane>=16 -> +8
            FragBF b;
            b.q[0] = *(const u32x4*)(bs + (ns * 16 + l15) * 40 + h * 8);
            b.q[1] = *(const u32x4*)(bs + (ns * 16 + l15) * 40 + h * 8 + 16);
            acc[ns] = __builtin_amdgcn_wmma_f32_16x16x32_bf16(
                false, a.v, false, b.v, (short)0, acc[ns], false, false);
        }
        __syncthreads();   // all reads of buf[cur] done before it is re-staged
    }

    // epilogue: C/D layout -> row = v + 8h, col = lane&15 (per 16-col subtile)
    #pragma unroll
    for (int ns = 0; ns < 4; ns++) {
        int col = n0 + ns * 16 + l15;
        float bv = bias[(size_t)e * Nd + col];
        #pragma unroll
        for (int v = 0; v < 8; v++) {
            int row = mBase + wid * 16 + v + 8 * h;
            float val = acc[ns][v] + bv;
            if (doGelu) {
                val = 0.5f * val * (1.0f + erff(val * 0.70710678118654752f));
                outH[(size_t)row * Nd + col] = f2bf(val);
            } else {
                outF[(size_t)row * Nd + col] = val;
            }
        }
    }
}

// ---------------- 6. combine: out[t] = g0*Y[s0] + g1*Y[s1] ----------------
__global__ void moe_combine(const float* __restrict__ Y,
                            const int* __restrict__ tslot,
                            const float* __restrict__ topg,
                            float* __restrict__ out) {
    int i = blockIdx.x * blockDim.x + threadIdx.x;
    if (i >= T_TOK * DDIM) return;
    int t = i / DDIM, d = i - t * DDIM;
    int s0 = tslot[t * 2 + 0], s1 = tslot[t * 2 + 1];
    out[i] = topg[t * 2 + 0] * Y[(size_t)s0 * DDIM + d]
           + topg[t * 2 + 1] * Y[(size_t)s1 * DDIM + d];
}

// ---------------- host launcher ----------------
extern "C" void kernel_launch(void* const* d_in, const int* in_sizes, int n_in,
                              void* d_out, int out_size, void* d_ws, size_t ws_size,
                              hipStream_t stream) {
    const float* x  = (const float*)d_in[0];
    const float* W1 = (const float*)d_in[1];
    const float* b1 = (const float*)d_in[2];
    const float* W2 = (const float*)d_in[3];
    const float* b2 = (const float*)d_in[4];
    const float* Wr = (const float*)d_in[5];
    const float* br = (const float*)d_in[6];
    float* out = (float*)d_out;

    char* ws = (char*)d_ws;
    size_t o = 0;
    auto take = [&](size_t bytes) -> char* {
        char* p = ws + o;
        o = (o + bytes + 255) & ~(size_t)255;
        return p;
    };
    float*          topg  = (float*)take((size_t)T_TOK * 2 * 4);
    int*            topi  = (int*)  take((size_t)T_TOK * 2 * 4);
    int*            meta  = (int*)  take(128 * 4);
    int*            p2t   = (int*)  take((size_t)MAXSLOTS * 4);
    int*            tslot = (int*)  take((size_t)T_TOK * 2 * 4);
    unsigned short* Xg    = (unsigned short*)take((size_t)MAXSLOTS * DDIM * 2);
    unsigned short* W1t   = (unsigned short*)take((size_t)NEXP * DDIM * IDIM * 2);
    unsigned short* W2t   = (unsigned short*)take((size_t)NEXP * DDIM * IDIM * 2);
    unsigned short* Hg    = (unsigned short*)take((size_t)MAXSLOTS * IDIM * 2);
    float*          Yb    = (float*)take((size_t)MAXSLOTS * DDIM * 4);
    (void)ws_size; (void)in_sizes; (void)n_in; (void)out_size;

    const int wconvBlocks = (NEXP * DDIM * IDIM) / 256;   // 73728
    // W1t[e][i][d] = W1[e][d][i]   (N0 = IDIM, K0 = DDIM)
    moe_convertT<<<wconvBlocks, 256, 0, stream>>>(W1, W1t, IDIM, DDIM);
    // W2t[e][d][i] = W2[e][i][d]   (N0 = DDIM, K0 = IDIM)
    moe_convertT<<<wconvBlocks, 256, 0, stream>>>(W2, W2t, DDIM, IDIM);

    moe_router <<<T_TOK, 256, 0, stream>>>(x, Wr, br, topg, topi);
    moe_build  <<<1, 256, 0, stream>>>(topi, meta, p2t, tslot);
    moe_gather <<<MAXSLOTS, 128, 0, stream>>>(x, p2t, meta, Xg);

    // GEMM1: [slots x 768] x [768 x 3072] -> gelu -> Hg (bf16)
    moe_gemm<<<dim3(IDIM / 64, MAXMBLK), 256, 0, stream>>>(
        Xg, W1t, b1, nullptr, Hg, meta, DDIM, IDIM, 1);
    // GEMM2: [slots x 3072] x [3072 x 768] -> + b2 -> Yb (f32)
    moe_gemm<<<dim3(DDIM / 64, MAXMBLK), 256, 0, stream>>>(
        Hg, W2t, b2, Yb, nullptr, meta, IDIM, DDIM, 0);

    moe_combine<<<(T_TOK * DDIM) / 256, 256, 0, stream>>>(Yb, tslot, topg, out);
}
